// SpatialCNN_23854248362243
// MI455X (gfx1250) — compile-verified, matrix operand order checked by reading, not run on hardware
//
#include <hip/hip_runtime.h>
#include <hip/hip_bf16.h>

// ---------------- problem constants (from reference) ----------------
#define BB  64
#define CC  128
#define HH  36
#define WW  100
#define KS  9
#define HWs (HH*WW)      // 3600
#define CHW (CC*HWs)     // 460800

typedef __attribute__((ext_vector_type(16))) __bf16 v16bf;
typedef __attribute__((ext_vector_type(8)))  float  v8f;

union Frag16 { v16bf v; uint4 q[2]; unsigned u[8]; };
union Acc8   { v8f   v; float f[8]; };

__device__ __forceinline__ unsigned short f2bf(float x) {
    unsigned u = __float_as_uint(x);
    unsigned r = u + 0x7fffu + ((u >> 16) & 1u);   // round-to-nearest-even
    return (unsigned short)(r >> 16);
}

// Pack 4 weight tensors (C_out, C_in, KS) f32 -> bf16 laid out [dir][k][co][ci]
// so A-fragments are two contiguous 16B chunks per lane.
__global__ __launch_bounds__(256)
void scnn_pack_weights(const float* __restrict__ wd, const float* __restrict__ wu,
                       const float* __restrict__ wr, const float* __restrict__ wl,
                       unsigned short* __restrict__ out)
{
    int idx = blockIdx.x * 256 + threadIdx.x;       // over 4*9*128*128
    if (idx >= 4 * KS * CC * CC) return;
    int ci = idx & (CC - 1);
    int t  = idx >> 7;
    int co = t & (CC - 1);
    t >>= 7;
    int k   = t % KS;
    int dir = t / KS;
    const float* w = (dir == 0) ? wd : (dir == 1) ? wu : (dir == 2) ? wr : wl;
    out[idx] = f2bf(w[(co * CC + ci) * KS + k]);
}

// One scan step: data[.., fixedOfs + pos*posStride] += relu( conv over prev line ).
// GEMM: M=128 (co), N=16 per block (b,pos pairs), K=1152 (ci x tap).
// Block = 8 waves; wave w owns co-tile w; B fragments shared via LDS.
__global__ __launch_bounds__(256)
void scnn_step(float* __restrict__ data,
               const unsigned short* __restrict__ wdir,   // [9][128][128] bf16
               int fixedOfs, int prevFixedOfs, int posStride, int posLen)
{
    __shared__ unsigned int sB[36 * 32 * 8];   // 36 fragments x 32 lanes x 32B = 36 KB
    const int tid = threadIdx.x;
    const int nTileBase = blockIdx.x * 16;

    // ---- stage B fragments (prev-line activations, f32 -> bf16) into LDS ----
    // fragment f = k*4 + ciBlock ; per-lane layout per ISA 7.12.2 (16-bit A/B).
    #pragma unroll 4
    for (int idx = tid; idx < 36 * 32 * 8; idx += 256) {
        int f   = idx >> 8;           // 0..35
        int rem = idx & 255;
        int l   = rem >> 3;           // lane 0..31
        int d   = rem & 7;            // dword 0..7 (2 bf16 each)
        int k   = f >> 2;             // tap 0..8
        int cb  = f & 3;              // ci block 0..3 (32 ci each)
        int nl  = l & 15;
        int kb  = (l < 16) ? 0 : 8;
        int K0  = (d < 4) ? (kb + 2 * d) : (kb + 16 + 2 * (d - 4));
        int n   = nTileBase + nl;
        int b   = n / posLen;
        int pos = n - b * posLen;
        int sp  = pos + k - (KS / 2);
        unsigned pack = 0u;
        if (sp >= 0 && sp < posLen) {                 // zero padding at borders
            int base = b * CHW + prevFixedOfs + sp * posStride;
            int ci0  = cb * 32 + K0;
            float f0 = data[base + ci0 * HWs];
            float f1 = data[base + (ci0 + 1) * HWs];
            pack = (unsigned)f2bf(f0) | ((unsigned)f2bf(f1) << 16);
        }
        sB[idx] = pack;
    }
    __syncthreads();

    // ---- per-wave WMMA accumulation: 9 taps x 4 ci-blocks ----
    const int wave = tid >> 5;          // co tile 0..7
    const int lane = tid & 31;
    const int m    = lane & 15;
    const int kb   = (lane < 16) ? 0 : 8;

    Acc8 acc;
    #pragma unroll
    for (int r = 0; r < 8; ++r) acc.f[r] = 0.0f;

    const int co = wave * 16 + m;
    for (int k = 0; k < KS; ++k) {
        #pragma unroll
        for (int cb = 0; cb < 4; ++cb) {
            // A fragment: weights, two contiguous 16B chunks (ci kb..kb+7, kb+16..kb+23)
            const uint4* wp = (const uint4*)(wdir + ((k * CC + co) * CC + cb * 32 + kb));
            Frag16 A;
            A.q[0] = wp[0];
            A.q[1] = wp[2];            // +16 bf16 = +32B
            // B fragment: from LDS (ds_load_b128 x2)
            Frag16 Bf;
            const uint4* p4 = (const uint4*)&sB[((k * 4 + cb) * 32 + lane) * 8];
            Bf.q[0] = p4[0];
            Bf.q[1] = p4[1];
            acc.v = __builtin_amdgcn_wmma_f32_16x16x32_bf16(
                        false, A.v, false, Bf.v, (short)0, acc.v, false, false);
        }
    }

    // ---- D layout: VGPR r -> M = r (+8 for upper lane half), N = lane&15 ----
    const int nl  = lane & 15;
    const int n   = nTileBase + nl;
    const int b   = n / posLen;
    const int pos = n - b * posLen;
    const int obase = b * CHW + fixedOfs + pos * posStride;
    const int mrow0 = (lane < 16) ? 0 : 8;
    #pragma unroll
    for (int r = 0; r < 8; ++r) {
        int coOut = wave * 16 + mrow0 + r;
        int a = obase + coOut * HWs;
        float v = acc.f[r];
        data[a] += (v > 0.0f) ? v : 0.0f;     // x + relu(conv)
    }
}

extern "C" void kernel_launch(void* const* d_in, const int* in_sizes, int n_in,
                              void* d_out, int out_size, void* d_ws, size_t ws_size,
                              hipStream_t stream)
{
    const float* x  = (const float*)d_in[0];
    const float* wd = (const float*)d_in[1];
    const float* wu = (const float*)d_in[2];
    const float* wr = (const float*)d_in[3];
    const float* wl = (const float*)d_in[4];
    float* out = (float*)d_out;
    unsigned short* wpack = (unsigned short*)d_ws;   // 4*9*128*128 bf16 = 1.18 MB

    // result tensor evolves in place in d_out
    hipMemcpyAsync(out, x, (size_t)BB * CHW * sizeof(float),
                   hipMemcpyDeviceToDevice, stream);

    const int wtot = 4 * KS * CC * CC;
    scnn_pack_weights<<<(wtot + 255) / 256, 256, 0, stream>>>(wd, wu, wr, wl, wpack);

    const int NtH = BB * WW / 16;   // 400 N-tiles for H-direction scans
    const int NtW = BB * HH / 16;   // 144 N-tiles for W-direction scans
    const unsigned short* pd = wpack + 0 * KS * CC * CC;
    const unsigned short* pu = wpack + 1 * KS * CC * CC;
    const unsigned short* pr = wpack + 2 * KS * CC * CC;
    const unsigned short* pl = wpack + 3 * KS * CC * CC;

    // down: h = 1..H-1 uses row h-1
    for (int h = 1; h < HH; ++h)
        scnn_step<<<NtH, 256, 0, stream>>>(out, pd, h * WW, (h - 1) * WW, 1, WW);
    // up: h = H-2..0 uses row h+1
    for (int h = HH - 2; h >= 0; --h)
        scnn_step<<<NtH, 256, 0, stream>>>(out, pu, h * WW, (h + 1) * WW, 1, WW);
    // right: w = 1..W-1 uses col w-1 (conv along H, element stride W)
    for (int w = 1; w < WW; ++w)
        scnn_step<<<NtW, 256, 0, stream>>>(out, pr, w, w - 1, WW, HH);
    // left: w = W-2..0 uses col w+1
    for (int w = WW - 2; w >= 0; --w)
        scnn_step<<<NtW, 256, 0, stream>>>(out, pl, w, w + 1, WW, HH);
}
